// BertSelfAttention_34797825032905
// MI455X (gfx1250) — matrix operand order repaired and checked by reading, not compiled, not run on hardware
//
#include <hip/hip_runtime.h>

// ---------------------------------------------------------------- types
typedef __bf16 bf16_t;
typedef __attribute__((ext_vector_type(4)))  __bf16 v4bf;
typedef __attribute__((ext_vector_type(8)))  __bf16 v8bf;
typedef __attribute__((ext_vector_type(16))) __bf16 v16bf;
typedef __attribute__((ext_vector_type(8)))  float  v8f;
typedef __attribute__((ext_vector_type(4)))  unsigned int u32x4;
typedef __attribute__((ext_vector_type(8)))  unsigned int u32x8;

// ---------------------------------------------------------------- problem
constexpr int Bn  = 4;
constexpr int Sn  = 2048;
constexpr int Hn  = 1024;
constexpr int NHn = 16;
constexpr int HDn = 64;

// workspace layout (bytes)
constexpr size_t XB_OFF = 0;                                   // X  bf16 [B*S, H]      16 MiB
constexpr size_t WT_OFF = (size_t)Bn*Sn*Hn*2;                  // Wt bf16 3x[H, H]       6 MiB
constexpr size_t Q_OFF  = WT_OFF + (size_t)3*Hn*Hn*2;          // Q  bf16 [B,NH,S,HD]   16 MiB
constexpr size_t KT_OFF = Q_OFF  + (size_t)Bn*NHn*Sn*HDn*2;    // Kt bf16 [B,NH,HD,S]   16 MiB
constexpr size_t V_OFF  = KT_OFF + (size_t)Bn*NHn*Sn*HDn*2;    // V  bf16 [B,NH,S,HD]   16 MiB

// ---------------------------------------------------------------- helpers
static __device__ __forceinline__ v16bf cat8(v8bf lo, v8bf hi) {
  return __builtin_shufflevector(lo, hi, 0,1,2,3,4,5,6,7,8,9,10,11,12,13,14,15);
}
// B-fragment (32x16, bf16): lane holds column n = lane%16; elements s<8: k = 8h+s, s>=8: k = 16+8h+(s-8)
// -> strided gather (row stride 64) from a [K, N] (transposed) operand; p points at column n.
static __device__ __forceinline__ v16bf load_bfrag(const bf16_t* p) {
  const int hh = (threadIdx.x >> 4) & 1;
  v16bf v;
  #pragma unroll
  for (int s = 0; s < 8; ++s) {
    v[s]     = p[(8 * hh + s) * 64];
    v[8 + s] = p[(16 + 8 * hh + s) * 64];
  }
  return v;
}
// A-fragment (16x32, bf16): lane (m = lane%16, h = lane/16):
//   VGPR 0..3 : k = 8h + {0..7},  VGPR 4..7 : k = 16 + 8h + {0..7}
static __device__ __forceinline__ v16bf load_afrag(const bf16_t* row, int kb, int h) {
  v8bf lo = *(const v8bf*)(row + kb + 8*h);
  v8bf hi = *(const v8bf*)(row + kb + 16 + 8*h);
  return cat8(lo, hi);
}
static __device__ __forceinline__ v8f wmma_bf16(v16bf a, v16bf b, v8f c) {
  return __builtin_amdgcn_wmma_f32_16x16x32_bf16(false, a, false, b, (short)0, c, false, false);
}
static __device__ __forceinline__ float hmax16(float x) {
  x = fmaxf(x, __shfl_xor(x, 1, 32));
  x = fmaxf(x, __shfl_xor(x, 2, 32));
  x = fmaxf(x, __shfl_xor(x, 4, 32));
  x = fmaxf(x, __shfl_xor(x, 8, 32));
  return x;
}
static __device__ __forceinline__ float hsum16(float x) {
  x += __shfl_xor(x, 1, 32);
  x += __shfl_xor(x, 2, 32);
  x += __shfl_xor(x, 4, 32);
  x += __shfl_xor(x, 8, 32);
  return x;
}

// ---------------------------------------------------------------- TDM: 2D tile Global -> LDS
// D# per CDNA5 ISA.  data_size = 4B units.
// group0: [1:0]=count=1, [63:32]=lds_addr, [120:64]=global_addr, [127:126]=type=2
// group1: [17:16]=data_size=2(4B), dim0/dim1 = tile bounds, tile_dim0/1, dim0_stride
static __device__ __forceinline__ uint32_t lds_addr_of(const void* p) {
  return (uint32_t)(uintptr_t)p;   // generic shared addr: low 32 bits = LDS byte offset
}
static __device__ __forceinline__ void tdm_load_2d(const void* gptr, uint32_t lds,
                                                   uint32_t rows, uint32_t cols4,
                                                   uint32_t stride4) {
  const uint64_t ga = (uint64_t)(uintptr_t)gptr;
  u32x4 g0;
  g0[0] = 1u;                                                   // count=1, user mode
  g0[1] = lds;
  g0[2] = (uint32_t)ga;
  g0[3] = ((uint32_t)(ga >> 32) & 0x01FFFFFFu) | 0x80000000u;   // addr hi | type=2
  u32x8 g1;
  g1[0] = 2u << 16;                                             // data_size = 4B
  g1[1] = (cols4 & 0xFFFFu) << 16;                              // tensor_dim0 lo16
  g1[2] = (cols4 >> 16) | ((rows & 0xFFFFu) << 16);             // dim0 hi | dim1 lo
  g1[3] = (rows >> 16) | (cols4 << 16);                         // dim1 hi | tile_dim0
  g1[4] = rows;                                                 // tile_dim1 (tile_dim2=0)
  g1[5] = stride4;                                              // tensor_dim0_stride lo32
  g1[6] = 0u;
  g1[7] = 0u;
  asm volatile("tensor_load_to_lds %0, %1" :: "s"(g0), "s"(g1) : "memory");
}

// ---------------------------------------------------------------- 1) f32 -> bf16 activations
__global__ __launch_bounds__(256) void cvt_x_kernel(const float* __restrict__ x,
                                                    bf16_t* __restrict__ xb) {
  int idx = blockIdx.x * blockDim.x + threadIdx.x;   // 4 elems per thread
  const float4 v = ((const float4*)x)[idx];
  v4bf o;
  o[0] = (bf16_t)v.x; o[1] = (bf16_t)v.y; o[2] = (bf16_t)v.z; o[3] = (bf16_t)v.w;
  ((v4bf*)xb)[idx] = o;
}

// ---------------------------------------------------------------- 2) f32 -> bf16 weights, transposed
__global__ __launch_bounds__(256) void cvt_w_kernel(const float* __restrict__ wq,
                                                    const float* __restrict__ wk,
                                                    const float* __restrict__ wv,
                                                    bf16_t* __restrict__ wt) {
  int idx = blockIdx.x * blockDim.x + threadIdx.x;   // 3 * H * H threads
  int z   = idx >> 20;
  int rem = idx & ((Hn * Hn) - 1);
  int o   = rem >> 10;        // output-feature row of W
  int hc  = rem & (Hn - 1);   // input-feature col (fastest -> coalesced read)
  const float* src = (z == 0) ? wq : ((z == 1) ? wk : wv);
  wt[(size_t)z * Hn * Hn + (size_t)hc * Hn + o] = (bf16_t)src[(size_t)o * Hn + hc];
}

// ---------------------------------------------------------------- 3) QKV projection GEMM (WMMA + TDM)
// grid (64, 16, 3) block 256.  block tile 128(M) x 64(N); wave tile 16 x 64.
// W tile (32k x 64n bf16 = 4 KB) staged via TDM, double buffered, shared by 8 waves.
__global__ __launch_bounds__(256) void proj_kernel(const bf16_t* __restrict__ xb,
                                                   const bf16_t* __restrict__ wt,
                                                   const float* __restrict__ bq,
                                                   const float* __restrict__ bk,
                                                   const float* __restrict__ bv,
                                                   bf16_t* __restrict__ q_ws,
                                                   bf16_t* __restrict__ kt_ws,
                                                   bf16_t* __restrict__ v_ws) {
  const int z    = blockIdx.z;
  const bf16_t* w = wt + (size_t)z * Hn * Hn;            // [H(k), H(o)] transposed
  const float* bias = (z == 0) ? bq : ((z == 1) ? bk : bv);

  const int wave = threadIdx.x >> 5;
  const int lane = threadIdx.x & 31;
  const int l = lane & 15, h = lane >> 4;

  const int mbase = blockIdx.x * 128 + wave * 16;
  const int nbase = blockIdx.y * 64;

  __shared__ __align__(16) bf16_t wtile[2][32 * 64];     // 2 x 4 KB double buffer

  v8f c0 = {}, c1 = {}, c2 = {}, c3 = {};
  const bf16_t* arow = xb + (size_t)(mbase + l) * Hn;
  const int kidx = l + 16 * h;                           // B-fragment lane k index

  // prologue: DMA first W tile, prefetch first A fragment
  if (wave == 0)
    tdm_load_2d(w + nbase, lds_addr_of(&wtile[0][0]), 32, 32, Hn / 2);
  v16bf a = load_afrag(arow, 0, h);

  int cur = 0;
  for (int ks = 0; ks < Hn / 32; ++ks) {
    if (wave == 0) __builtin_amdgcn_s_wait_tensorcnt(0); // buf[cur] landed in LDS
    __syncthreads();                                     // visible to all; prev reads done
    if (wave == 0 && ks + 1 < Hn / 32)                   // DMA next tile into other buffer
      tdm_load_2d(w + (size_t)(ks + 1) * 32 * Hn + nbase,
                  lds_addr_of(&wtile[cur ^ 1][0]), 32, 32, Hn / 2);
    v16bf a_nxt = (ks + 1 < Hn / 32) ? load_afrag(arow, (ks + 1) * 32, h) : a;

    const bf16_t* wl = &wtile[cur][l];
    c0 = wmma_bf16(a, load_bfrag(wl +  0), c0);
    c1 = wmma_bf16(a, load_bfrag(wl + 16), c1);
    c2 = wmma_bf16(a, load_bfrag(wl + 32), c2);
    c3 = wmma_bf16(a, load_bfrag(wl + 48), c3);

    a = a_nxt;
    cur ^= 1;
  }

  const int bb = mbase >> 11;          // batch (uniform: mbase 16-aligned)
  const int sl = mbase & (Sn - 1);     // seq row base within batch

  v8f cs[4] = {c0, c1, c2, c3};
  #pragma unroll
  for (int t = 0; t < 4; ++t) {
    const int ob    = nbase + 16 * t;
    const int head  = ob >> 6;
    const int dbase = ob & 63;
    const int d     = dbase + l;
    const float bval = bias[ob + l];
    if (z == 1) {
      // K transposed layout [B,NH,HD,S]: consecutive r -> consecutive s: pack b128
      v8bf pk;
      #pragma unroll
      for (int r = 0; r < 8; ++r) pk[r] = (bf16_t)(cs[t][r] + bval);
      bf16_t* kp = kt_ws + (((size_t)(bb * NHn + head) * HDn + d) * Sn) + sl + 8 * h;
      *(v8bf*)kp = pk;
    } else {
      bf16_t* dst = ((z == 0) ? q_ws : v_ws) + (size_t)(bb * NHn + head) * Sn * HDn;
      #pragma unroll
      for (int r = 0; r < 8; ++r)
        dst[(size_t)(sl + r + 8 * h) * HDn + d] = (bf16_t)(cs[t][r] + bval);
    }
  }
}

// ---------------------------------------------------------------- 4) flash attention (WMMA + TDM)
// grid (S/128, B*NH) block 256; each wave owns 16 query rows.
// Kt tile (64d x 64s) and V tile (64s x 64d) staged via TDM, double buffered,
// shared by all 8 waves (8x traffic reduction vs per-wave global loads).
__global__ __launch_bounds__(256) void attn_kernel(const bf16_t* __restrict__ q_ws,
                                                   const bf16_t* __restrict__ kt_ws,
                                                   const bf16_t* __restrict__ v_ws,
                                                   const float* __restrict__ mask,
                                                   float* __restrict__ out) {
  const int bh   = blockIdx.y;
  const int bb   = bh >> 4;
  const int head = bh & 15;
  const int wave = threadIdx.x >> 5;
  const int lane = threadIdx.x & 31;
  const int l = lane & 15, h = lane >> 4;

  const int q0 = blockIdx.x * 128 + wave * 16;

  const bf16_t* qrow = q_ws + ((size_t)bh * Sn + q0 + l) * HDn;
  const v16bf qa0 = load_afrag(qrow, 0, h);
  const v16bf qa1 = load_afrag(qrow, 32, h);

  const bf16_t* ktb = kt_ws + (size_t)bh * HDn * Sn;
  const bf16_t* vb  = v_ws  + (size_t)bh * Sn * HDn;
  const float*  mrw = mask + (size_t)bb * Sn;

  __shared__ __align__(16) bf16_t ktile[2][64 * 64];   // 2 x 8 KB
  __shared__ __align__(16) bf16_t vtile[2][64 * 64];   // 2 x 8 KB
  __shared__ __align__(16) bf16_t plds[8][16 * 72];    // per-wave P tile, stride 72

  bf16_t* pw = &plds[wave][0];

  const float SC  = 0.125f * 1.44269504088896f;        // 1/sqrt(HD) * log2(e)
  const float L2E = 1.44269504088896f;

  float mst[8], lst[8];
  v8f o0 = {}, o1 = {}, o2 = {}, o3 = {};
  #pragma unroll
  for (int r = 0; r < 8; ++r) { mst[r] = -3.0e38f; lst[r] = 0.0f; }

  const int kidx = l + 16 * h;                         // B-fragment lane k index
  constexpr int NKB = Sn / 64;

  // prologue: DMA first K/V tiles
  if (wave == 0) {
    tdm_load_2d(ktb, lds_addr_of(&ktile[0][0]), 64, 32, Sn / 2);   // Kt: 64 d-rows x 64 s
    tdm_load_2d(vb,  lds_addr_of(&vtile[0][0]), 64, 32, HDn / 2);  // V : 64 s-rows x 64 d
  }

  int cur = 0;
  for (int ikb = 0; ikb < NKB; ++ikb) {
    const int kb = ikb * 64;
    if (wave == 0) __builtin_amdgcn_s_wait_tensorcnt(0);
    __syncthreads();
    if (wave == 0 && ikb + 1 < NKB) {
      tdm_load_2d(ktb + (kb + 64), lds_addr_of(&ktile[cur ^ 1][0]), 64, 32, Sn / 2);
      tdm_load_2d(vb + (size_t)(kb + 64) * HDn, lds_addr_of(&vtile[cur ^ 1][0]), 64, 32, HDn / 2);
    }

    // ---- scores: Q(16x64) x Kt -> 4 tiles of 16x16 (f32)
    v8f s[4];
    #pragma unroll
    for (int t = 0; t < 4; ++t) {
      v8f acc = {};
      acc = wmma_bf16(qa0, load_bfrag(&ktile[cur][(0)  * 64 + 16 * t + l]), acc);
      acc = wmma_bf16(qa1, load_bfrag(&ktile[cur][(32) * 64 + 16 * t + l]), acc);
      const float mk = mrw[kb + 16 * t + l] * L2E;     // additive mask (col n = l)
      #pragma unroll
      for (int r = 0; r < 8; ++r) s[t][r] = fmaf(acc[r], SC, mk);
    }
    // ---- online softmax (row m = r + 8h lives on the 16-lane half, per VGPR r)
    float alpha[8], nm[8], rs[8];
    #pragma unroll
    for (int r = 0; r < 8; ++r) {
      float x = fmaxf(fmaxf(s[0][r], s[1][r]), fmaxf(s[2][r], s[3][r]));
      x = hmax16(x);
      nm[r]    = fmaxf(mst[r], x);
      alpha[r] = exp2f(mst[r] - nm[r]);
      rs[r]    = 0.0f;
    }
    #pragma unroll
    for (int t = 0; t < 4; ++t)
      #pragma unroll
      for (int r = 0; r < 8; ++r) {
        float p = exp2f(s[t][r] - nm[r]);
        rs[r] += p;
        pw[(r + 8 * h) * 72 + 16 * t + l] = (bf16_t)p;   // C-layout -> LDS row-major
      }
    #pragma unroll
    for (int r = 0; r < 8; ++r) {
      lst[r] = lst[r] * alpha[r] + hsum16(rs[r]);
      mst[r] = nm[r];
      o0[r] *= alpha[r]; o1[r] *= alpha[r]; o2[r] *= alpha[r]; o3[r] *= alpha[r];
    }
    asm volatile("s_wait_dscnt 0" ::: "memory");         // wave-private P tile store->load
    // ---- PV: P(16x64) x V -> accumulate O (16x64)
    #pragma unroll
    for (int kv = 0; kv < 2; ++kv) {
      v16bf pa = load_afrag(pw + l * 72, 32 * kv, h);    // ds_load_b128 x2
      const bf16_t* vrow = &vtile[cur][(32 * kv) * 64 + l];
      o0 = wmma_bf16(pa, load_bfrag(vrow +  0), o0);
      o1 = wmma_bf16(pa, load_bfrag(vrow + 16), o1);
      o2 = wmma_bf16(pa, load_bfrag(vrow + 32), o2);
      o3 = wmma_bf16(pa, load_bfrag(vrow + 48), o3);
    }
    cur ^= 1;
  }

  // ---- epilogue: O / l, scatter to [B,S,H] f32
  float rl[8];
  #pragma unroll
  for (int r = 0; r < 8; ++r) rl[r] = 1.0f / lst[r];
  v8f os[4] = {o0, o1, o2, o3};
  #pragma unroll
  for (int dt = 0; dt < 4; ++dt)
    #pragma unroll
    for (int r = 0; r < 8; ++r) {
      const size_t srow = (size_t)bb * Sn + q0 + r + 8 * h;
      out[srow * Hn + head * 64 + dt * 16 + l] = os[dt][r] * rl[r];
    }
}

// ---------------------------------------------------------------- launch
extern "C" void kernel_launch(void* const* d_in, const int* in_sizes, int n_in,
                              void* d_out, int out_size, void* d_ws, size_t ws_size,
                              hipStream_t stream) {
  const float* hs   = (const float*)d_in[0];
  const float* mask = (const float*)d_in[1];
  const float* Wq   = (const float*)d_in[2];
  const float* bq   = (const float*)d_in[3];
  const float* Wk   = (const float*)d_in[4];
  const float* bk   = (const float*)d_in[5];
  const float* Wv   = (const float*)d_in[6];
  const float* bv   = (const float*)d_in[7];
  float* out = (float*)d_out;

  char* ws = (char*)d_ws;
  bf16_t* xb    = (bf16_t*)(ws + XB_OFF);
  bf16_t* wt    = (bf16_t*)(ws + WT_OFF);
  bf16_t* q_ws  = (bf16_t*)(ws + Q_OFF);
  bf16_t* kt_ws = (bf16_t*)(ws + KT_OFF);
  bf16_t* v_ws  = (bf16_t*)(ws + V_OFF);

  cvt_x_kernel<<<(Bn * Sn * Hn) / 4 / 256, 256, 0, stream>>>(hs, xb);
  cvt_w_kernel<<<(3 * Hn * Hn) / 256, 256, 0, stream>>>(Wq, Wk, Wv, wt);
  proj_kernel<<<dim3((Bn * Sn) / 128, Hn / 64, 3), 256, 0, stream>>>(
      xb, wt, bq, bk, bv, q_ws, kt_ws, v_ws);
  attn_kernel<<<dim3(Sn / 128, Bn * NHn), 256, 0, stream>>>(
      q_ws, kt_ws, v_ws, mask, out);
}
